// NonLocalBlock_29987461660808
// MI455X (gfx1250) — matrix operand order repaired and checked by reading, not compiled, not run on hardware
//
#include <hip/hip_runtime.h>
#include <hip/hip_bf16.h>

// ---------------------------------------------------------------------------
// NonLocal block for MI455X (gfx1250, wave32).
// All GEMMs: LDS-staged, register-blocked v_wmma_f32_16x16x32_bf16.
// Block tile 128(M) x 64(N) x 32(K); 8 waves; each wave: 32x32 (2x2 WMMA).
// A-tile (and col-K B-tile) staged with GLOBAL_LOAD_ASYNC_TO_LDS_B128
// (ASYNCcnt domain); transposing stages use ds_store.
// ---------------------------------------------------------------------------

typedef __attribute__((ext_vector_type(16))) __bf16 v16bf;
typedef __attribute__((ext_vector_type(8)))  float  v8f;

#define BN_EPS 1e-5f

static constexpr int Bn = 16;     // batch
static constexpr int C  = 512;    // channels
static constexpr int Ci = 256;    // inner channels
static constexpr int N  = 4096;   // H*W

static constexpr int LDP = 40;    // padded LDS row (32 + 8 bf16) -> 80B stride

union frag_u { v16bf v; uint4 q[2]; };

__device__ __forceinline__ v8f wmma_bf16(v16bf a, v16bf b, v8f c) {
  // (neg_a, A, neg_b, B, c_mod, C, reuse_a, reuse_b)
  return __builtin_amdgcn_wmma_f32_16x16x32_bf16(false, a, false, b, (short)0, c,
                                                 false, false);
}

// async 16B global -> LDS copy (per lane), tracked by ASYNCcnt
__device__ __forceinline__ void async_copy_b128(const void* lds_dst, const void* gsrc) {
  const unsigned       l = (unsigned)(size_t)lds_dst;      // wave LDS byte offset
  const unsigned long long g = (unsigned long long)(size_t)gsrc;
  asm volatile("global_load_async_to_lds_b128 %0, %1, off"
               :: "v"(l), "v"(g) : "memory");
}
__device__ __forceinline__ void async_wait0() {
  asm volatile("s_wait_asynccnt 0x0" ::: "memory");
}

// ---------------------------------------------------------------------------
// GEMM core: C_tile += A(row-major MxK, bf16) * B (source layout by BS).
//   BS=0: B bf16, element (k,n) at B[k*ldb + n]         (row-K)
//   BS=1: B bf16, element (k,n) at B[n*ldb + k]         (col-K, contiguous K)
//   BS=2: B f32,  element (k,n) at B[k*ldb + n]         (row-K, cvt to bf16)
// Accumulators: acc[mi][ni] -> 16x16 tile at (wm+16mi, wn+16ni).
// ---------------------------------------------------------------------------
template <int BS>
__device__ __forceinline__ void gemm_block(
    const __bf16* __restrict__ A, int lda,
    const void*  __restrict__ Bsrc, int ldb,
    int row0, int col0, int K, v8f (&acc)[2][2]) {
  __shared__ __bf16 sA[128][LDP];   // [m][k]
  __shared__ __bf16 sB[64][LDP];    // [n][k]  (K contiguous for b128 frag reads)

  const int t    = threadIdx.x;
  const int lane = t & 31;
  const int wave = t >> 5;
  const int wm   = (wave >> 1) * 32;      // wave M offset in block tile
  const int wn   = (wave & 1)  * 32;      // wave N offset

  // staging index precompute
  const int a_row   = t >> 2;             // 0..63 (+64 for second row)
  const int a_koff  = (t & 3) * 8;        // 0,8,16,24
  const int b_k     = t >> 3;             // 0..31   (row-K sources)
  const int b_noff  = (t & 7) * 8;        // 0..56
  const int bc_n    = t >> 2;             // 0..63   (col-K source)
  const int bc_koff = (t & 3) * 8;

  // fragment index precompute
  const int ar  = lane & 15;
  const int akb = (lane >> 4) << 3;       // 0 / 8
  const int bn  = lane & 15;
  const int bkb = (lane >> 4) << 3;       // 0 / 8

  for (int k0 = 0; k0 < K; k0 += 32) {
    __syncthreads();   // previous iteration's fragment reads done before overwrite
    // ---- stage A tile (128 x 32) via async DMA to LDS ----
    async_copy_b128(&sA[a_row][a_koff],
                    A + (size_t)(row0 + a_row) * lda + k0 + a_koff);
    async_copy_b128(&sA[a_row + 64][a_koff],
                    A + (size_t)(row0 + a_row + 64) * lda + k0 + a_koff);
    // ---- stage B tile (32 x 64) into [n][k] layout ----
    if (BS == 0) {                       // bf16 row-K: transpose on store
      const __bf16* Bb = (const __bf16*)Bsrc;
      const uint4 q = *(const uint4*)(Bb + (size_t)(k0 + b_k) * ldb + col0 + b_noff);
      const __bf16* e = (const __bf16*)&q;
#pragma unroll
      for (int i = 0; i < 8; ++i) sB[b_noff + i][b_k] = e[i];
    } else if (BS == 1) {                // bf16 col-K: straight async DMA copy
      const __bf16* Bb = (const __bf16*)Bsrc;
      async_copy_b128(&sB[bc_n][bc_koff],
                      Bb + (size_t)(col0 + bc_n) * ldb + k0 + bc_koff);
    } else {                             // f32 row-K: cvt + transpose on store
      const float* Bf = (const float*)Bsrc;
      const float4 q0 = *(const float4*)(Bf + (size_t)(k0 + b_k) * ldb + col0 + b_noff);
      const float4 q1 = *(const float4*)(Bf + (size_t)(k0 + b_k) * ldb + col0 + b_noff + 4);
      const __bf16 e[8] = {(__bf16)q0.x, (__bf16)q0.y, (__bf16)q0.z, (__bf16)q0.w,
                           (__bf16)q1.x, (__bf16)q1.y, (__bf16)q1.z, (__bf16)q1.w};
#pragma unroll
      for (int i = 0; i < 8; ++i) sB[b_noff + i][b_k] = e[i];
    }
    // prefetch next K slice while this tile is consumed (global_prefetch_b8)
    if (k0 + 32 < K) {
      __builtin_prefetch(A + (size_t)(row0 + a_row) * lda + k0 + 32 + a_koff, 0, 1);
      if (BS == 2)
        __builtin_prefetch((const float*)Bsrc + (size_t)(k0 + 32 + b_k) * ldb + col0 + b_noff, 0, 1);
      else if (BS == 0)
        __builtin_prefetch((const __bf16*)Bsrc + (size_t)(k0 + 32 + b_k) * ldb + col0 + b_noff, 0, 1);
      else
        __builtin_prefetch((const __bf16*)Bsrc + (size_t)(col0 + bc_n) * ldb + k0 + 32 + bc_koff, 0, 1);
    }
    async_wait0();     // our async DMAs landed in LDS (compiler handles DScnt)
    __syncthreads();
    // ---- fragments (aligned b128 LDS reads) + 2x2 WMMA ----
    v16bf afrag[2], bfrag[2];
#pragma unroll
    for (int i = 0; i < 2; ++i) {
      frag_u ua;
      ua.q[0] = *(const uint4*)&sA[wm + 16 * i + ar][akb];
      ua.q[1] = *(const uint4*)&sA[wm + 16 * i + ar][akb + 16];
      afrag[i] = ua.v;
      frag_u ub;
      ub.q[0] = *(const uint4*)&sB[wn + 16 * i + bn][bkb];
      ub.q[1] = *(const uint4*)&sB[wn + 16 * i + bn][bkb + 16];
      bfrag[i] = ub.v;
    }
#pragma unroll
    for (int mi = 0; mi < 2; ++mi)
#pragma unroll
      for (int ni = 0; ni < 2; ++ni)
        acc[mi][ni] = wmma_bf16(afrag[mi], bfrag[ni], acc[mi][ni]);
  }
}

// ---------------------------------------------------------------------------
// Kernel 1: weight / bias prep (fp32 -> bf16, fuse g/theta/phi)
// ---------------------------------------------------------------------------
__global__ __launch_bounds__(256) void k_prep(
    const float* __restrict__ g_w, const float* __restrict__ th_w,
    const float* __restrict__ ph_w, const float* __restrict__ g_b,
    const float* __restrict__ th_b, const float* __restrict__ ph_b,
    const float* __restrict__ W_w,
    __bf16* __restrict__ Wqkv, __bf16* __restrict__ Wz, float* __restrict__ bqkv) {
  const int idx = blockIdx.x * 256 + threadIdx.x;
  if (idx < 768 * 512) {                       // fused (3*Ci) x C weight
    const int r = idx >> 9, k = idx & 511;
    const float* src = (r < 256) ? g_w : ((r < 512) ? th_w : ph_w);
    Wqkv[idx] = (__bf16)src[(r & 255) * 512 + k];
  }
  if (idx < 512 * 256) Wz[idx] = (__bf16)W_w[idx];   // C x Ci
  if (idx < 768) {
    const float* src = (idx < 256) ? g_b : ((idx < 512) ? th_b : ph_b);
    bqkv[idx] = src[idx & 255];
  }
}

// ---------------------------------------------------------------------------
// Kernel 2: fused g/theta/phi projections.
// out(768 x 4096) = Wqkv(768 x 512) * x_b(512 x 4096); gtp: [which][b][Ci][N]
// ---------------------------------------------------------------------------
__global__ __launch_bounds__(256) void k_gemm_qkv(
    const float* __restrict__ x, const __bf16* __restrict__ Wqkv,
    const float* __restrict__ bqkv, __bf16* __restrict__ gtp) {
  const int lane = threadIdx.x & 31;
  const int wave = threadIdx.x >> 5;
  const int b    = blockIdx.z;
  const int col0 = blockIdx.x * 64;
  const int row0 = blockIdx.y * 128;
  const float* xb = x + (size_t)b * C * N;

  v8f acc[2][2] = {};
  gemm_block<2>(Wqkv, C, xb, N, row0, col0, C, acc);

  const int wm = (wave >> 1) * 32, wn = (wave & 1) * 32;
#pragma unroll
  for (int mi = 0; mi < 2; ++mi) {
    const int rbase = row0 + wm + 16 * mi + ((lane >> 4) << 3);
#pragma unroll
    for (int ni = 0; ni < 2; ++ni) {
      const int n = col0 + wn + 16 * ni + (lane & 15);
#pragma unroll
      for (int v = 0; v < 8; ++v) {
        const int gr = rbase + v;
        const int which = gr >> 8;
        const int cr = gr & 255;
        const float val = acc[mi][ni][v] + bqkv[gr];
        gtp[((size_t)(which * Bn + b) * Ci + cr) * N + n] = (__bf16)val;
      }
    }
  }
}

// ---------------------------------------------------------------------------
// Kernel 3: f[b] = theta_b (256 x 4096) * phi_b^T -> (256 x 256), K = 4096.
// ---------------------------------------------------------------------------
__global__ __launch_bounds__(256) void k_gemm_f(
    const __bf16* __restrict__ gtp, float* __restrict__ fscore) {
  const int lane = threadIdx.x & 31;
  const int wave = threadIdx.x >> 5;
  const int b    = blockIdx.z;
  const int col0 = blockIdx.x * 64;
  const int row0 = blockIdx.y * 128;
  const __bf16* theta = gtp + (size_t)(1 * Bn + b) * Ci * N;
  const __bf16* phi   = gtp + (size_t)(2 * Bn + b) * Ci * N;

  v8f acc[2][2] = {};
  gemm_block<1>(theta, N, phi, N, row0, col0, N, acc);

  const int wm = (wave >> 1) * 32, wn = (wave & 1) * 32;
#pragma unroll
  for (int mi = 0; mi < 2; ++mi) {
    const int rbase = row0 + wm + 16 * mi + ((lane >> 4) << 3);
#pragma unroll
    for (int ni = 0; ni < 2; ++ni) {
      const int n = col0 + wn + 16 * ni + (lane & 15);
#pragma unroll
      for (int v = 0; v < 8; ++v)
        fscore[((size_t)b * Ci + (rbase + v)) * Ci + n] = acc[mi][ni][v];
    }
  }
}

// ---------------------------------------------------------------------------
// Kernel 4: row softmax over 256-wide rows; P stored bf16.
// ---------------------------------------------------------------------------
__global__ __launch_bounds__(256) void k_softmax(
    const float* __restrict__ fscore, __bf16* __restrict__ P) {
  __shared__ float red[256];
  const int row = blockIdx.x;                        // b*256 + c
  const int t   = threadIdx.x;
  const float v = fscore[(size_t)row * Ci + t];
  red[t] = v; __syncthreads();
  for (int s = 128; s > 0; s >>= 1) {
    if (t < s) red[t] = fmaxf(red[t], red[t + s]);
    __syncthreads();
  }
  const float mx = red[0]; __syncthreads();
  const float ev = __expf(v - mx);
  red[t] = ev; __syncthreads();
  for (int s = 128; s > 0; s >>= 1) {
    if (t < s) red[t] += red[t + s];
    __syncthreads();
  }
  P[(size_t)row * Ci + t] = (__bf16)(ev / red[0]);
}

// ---------------------------------------------------------------------------
// Kernel 5: y[b] = P[b] (256 x 256) * g_b (256 x 4096), K = 256.
// ---------------------------------------------------------------------------
__global__ __launch_bounds__(256) void k_gemm_y(
    const __bf16* __restrict__ P, const __bf16* __restrict__ gtp,
    __bf16* __restrict__ y) {
  const int lane = threadIdx.x & 31;
  const int wave = threadIdx.x >> 5;
  const int b    = blockIdx.z;
  const int col0 = blockIdx.x * 64;
  const int row0 = blockIdx.y * 128;
  const __bf16* Pb = P + (size_t)b * Ci * Ci;
  const __bf16* gb = gtp + (size_t)(0 * Bn + b) * Ci * N;

  v8f acc[2][2] = {};
  gemm_block<0>(Pb, Ci, gb, N, row0, col0, Ci, acc);

  const int wm = (wave >> 1) * 32, wn = (wave & 1) * 32;
#pragma unroll
  for (int mi = 0; mi < 2; ++mi) {
    const int rbase = row0 + wm + 16 * mi + ((lane >> 4) << 3);
#pragma unroll
    for (int ni = 0; ni < 2; ++ni) {
      const int n = col0 + wn + 16 * ni + (lane & 15);
#pragma unroll
      for (int v = 0; v < 8; ++v)
        y[((size_t)b * Ci + (rbase + v)) * N + n] = (__bf16)acc[mi][ni][v];
    }
  }
}

// ---------------------------------------------------------------------------
// Kernel 6: z[b] = Wz (512 x 256) * y[b] (256 x 4096), K = 256,
// fused per-channel sum / sum-of-squares (shuffle reduce + float atomics).
// ---------------------------------------------------------------------------
__global__ __launch_bounds__(256) void k_gemm_z(
    const __bf16* __restrict__ Wz, const __bf16* __restrict__ y,
    const float* __restrict__ Wb, float* __restrict__ z,
    float* __restrict__ sums) {
  const int lane = threadIdx.x & 31;
  const int wave = threadIdx.x >> 5;
  const int b    = blockIdx.z;
  const int col0 = blockIdx.x * 64;
  const int row0 = blockIdx.y * 128;
  const __bf16* yb = y + (size_t)b * Ci * N;
  float* zb = z + (size_t)b * C * N;

  v8f acc[2][2] = {};
  gemm_block<0>(Wz, Ci, yb, N, row0, col0, Ci, acc);

  const int wm = (wave >> 1) * 32, wn = (wave & 1) * 32;
#pragma unroll
  for (int mi = 0; mi < 2; ++mi) {
    const int rbase = row0 + wm + 16 * mi + ((lane >> 4) << 3);
#pragma unroll
    for (int ni = 0; ni < 2; ++ni) {
      const int n = col0 + wn + 16 * ni + (lane & 15);
#pragma unroll
      for (int v = 0; v < 8; ++v) {
        const int r = rbase + v;
        const float zv = acc[mi][ni][v] + Wb[r];
        zb[(size_t)r * N + n] = zv;
        float s = zv, sq = zv * zv;
#pragma unroll
        for (int m = 1; m < 16; m <<= 1) {
          s  += __shfl_xor(s, m, 32);
          sq += __shfl_xor(sq, m, 32);
        }
        if ((lane & 15) == 0) {
          atomicAdd(&sums[r], s);
          atomicAdd(&sums[C + r], sq);
        }
      }
    }
  }
}

// ---------------------------------------------------------------------------
// Kernel 7: BatchNorm (batch stats) + residual, float4 vectorized.
// ---------------------------------------------------------------------------
__global__ __launch_bounds__(256) void k_bn_res(
    const float* __restrict__ z, const float* __restrict__ x,
    const float* __restrict__ sums, const float* __restrict__ gamma,
    const float* __restrict__ beta, float* __restrict__ out) {
  const size_t i4 = (size_t)blockIdx.x * 256 + threadIdx.x;
  const size_t e  = i4 * 4;
  const int c = (int)((e >> 12) & (C - 1));          // (e / 4096) % 512
  const float inv  = 1.0f / (float)((size_t)Bn * N);
  const float mean = sums[c] * inv;
  const float var  = sums[C + c] * inv - mean * mean;
  const float sc   = gamma[c] * rsqrtf(var + BN_EPS);
  const float sh   = beta[c] - mean * sc;
  const float4 zv = *(const float4*)(z + e);
  const float4 xv = *(const float4*)(x + e);
  float4 o;
  o.x = zv.x * sc + sh + xv.x;
  o.y = zv.y * sc + sh + xv.y;
  o.z = zv.z * sc + sh + xv.z;
  o.w = zv.w * sc + sh + xv.w;
  *(float4*)(out + e) = o;
}

// ---------------------------------------------------------------------------
// Host launcher
// ---------------------------------------------------------------------------
extern "C" void kernel_launch(void* const* d_in, const int* in_sizes, int n_in,
                              void* d_out, int out_size, void* d_ws, size_t ws_size,
                              hipStream_t stream) {
  const float* x     = (const float*)d_in[0];
  const float* g_w   = (const float*)d_in[1];
  const float* g_b   = (const float*)d_in[2];
  const float* th_w  = (const float*)d_in[3];
  const float* th_b  = (const float*)d_in[4];
  const float* ph_w  = (const float*)d_in[5];
  const float* ph_b  = (const float*)d_in[6];
  const float* W_w   = (const float*)d_in[7];
  const float* W_b   = (const float*)d_in[8];
  const float* gamma = (const float*)d_in[9];
  const float* beta  = (const float*)d_in[10];
  float* out = (float*)d_out;

  // workspace carve-up (bytes)
  char* ws = (char*)d_ws;
  __bf16* gtp    = (__bf16*)(ws);                          // 3*B*Ci*N bf16 = 96 MB
  __bf16* yb     = (__bf16*)(ws + 100663296);              // B*Ci*N bf16  = 32 MB
  float*  z      = (float*) (ws + 134217728);              // B*C*N  f32   = 128 MB
  float*  fscore = (float*) (ws + 268435456);              // B*Ci*Ci f32  = 4 MB
  __bf16* P      = (__bf16*)(ws + 272629760);              // B*Ci*Ci bf16 = 2 MB
  __bf16* Wqkv   = (__bf16*)(ws + 274726912);              // 768*512 bf16
  __bf16* Wz     = (__bf16*)(ws + 275513344);              // 512*256 bf16
  float*  bqkv   = (float*) (ws + 275775488);              // 768 f32
  float*  sums   = (float*) (ws + 275778560);              // 2*512 f32

  dim3 blk(256, 1, 1);

  // 1. weights -> bf16
  k_prep<<<dim3((768 * 512) / 256, 1, 1), blk, 0, stream>>>(
      g_w, th_w, ph_w, g_b, th_b, ph_b, W_w, Wqkv, Wz, bqkv);

  // 2. fused g/theta/phi projection: (768x512)x(512x4096) per batch
  k_gemm_qkv<<<dim3(N / 64, 768 / 128, Bn), blk, 0, stream>>>(x, Wqkv, bqkv, gtp);

  // 3. f = theta * phi^T   (256x256, K=4096)
  k_gemm_f<<<dim3(Ci / 64, Ci / 128, Bn), blk, 0, stream>>>(gtp, fscore);

  // 4. softmax rows -> P (bf16)
  k_softmax<<<dim3(Bn * Ci, 1, 1), blk, 0, stream>>>(fscore, P);

  // 5. y = P * g          (256x4096, K=256)
  k_gemm_y<<<dim3(N / 64, Ci / 128, Bn), blk, 0, stream>>>(P, gtp, yb);

  // 6. z = W * y + fused BN statistics
  hipMemsetAsync(sums, 0, 2 * C * sizeof(float), stream);
  k_gemm_z<<<dim3(N / 64, C / 128, Bn), blk, 0, stream>>>(Wz, yb, W_b, z, sums);

  // 7. BatchNorm + residual
  k_bn_res<<<dim3((Bn * C * N) / (4 * 256), 1, 1), blk, 0, stream>>>(
      z, x, sums, gamma, beta, out);
}
